// SDGNN_7584912245437
// MI455X (gfx1250) — matrix operand order repaired and checked by reading, not compiled, not run on hardware
//
#include <hip/hip_runtime.h>

#define NN 100000
#define NE 1600000
#define NG 1024
#define HID 128

typedef __attribute__((ext_vector_type(16))) __bf16 v16bf;
typedef __attribute__((ext_vector_type(2)))  __bf16 v2bf;
typedef __attribute__((ext_vector_type(8)))  float  v8f;

union BfTile {
    uint4 u[2];
    v16bf v;
};

// hardware f32->bf16 (RNE) converts
__device__ __forceinline__ unsigned pk2bf(float a, float b) {
    v2bf t;
    t[0] = (__bf16)a;
    t[1] = (__bf16)b;
    return __builtin_bit_cast(unsigned, t);
}
__device__ __forceinline__ unsigned short f2bf(float f) {
    return __builtin_bit_cast(unsigned short, (__bf16)f);
}

__device__ __forceinline__ v8f wmma_bf16(v16bf a, v16bf b, v8f c) {
    return __builtin_amdgcn_wmma_f32_16x16x32_bf16(false, a, false, b, (short)0, c, false, false);
}

// One MLP layer over a 32-row tile: each wave owns a 16-col N chunk and issues
// two WMMAs per K-step (rows 0-15, rows 16-31) reusing one B tile in registers.
// sA: LDS bf16 activations [32][rowStride]; Bpl pre-offset to wave/lane.
template <int NKT>
__device__ __forceinline__ void mlp_layer2(const unsigned short* sA, int rowStride,
                                           const unsigned short* Bpl, int lane,
                                           v8f& acc0, v8f& acc1) {
    const int arow = lane & 15;
    const int koff = (lane >= 16) ? 16 : 0;  // contiguous k order (same as B pack): lanes 16-31 hold K+16
    const unsigned short* aBase0 = sA + arow * rowStride + koff;
    const unsigned short* aBase1 = sA + (arow + 16) * rowStride + koff;
#pragma unroll
    for (int kt = 0; kt < NKT; ++kt) {
        BfTile a0, a1, b;
        const uint4* pA0 = (const uint4*)(aBase0 + kt * 32);
        const uint4* pA1 = (const uint4*)(aBase1 + kt * 32);
        a0.u[0] = pA0[0]; a0.u[1] = pA0[1];
        a1.u[0] = pA1[0]; a1.u[1] = pA1[1];
        const uint4* pB = (const uint4*)(Bpl + kt * (8 * 512));
        b.u[0] = pB[0]; b.u[1] = pB[1];
        acc0 = wmma_bf16(a0.v, b.v, acc0);
        acc1 = wmma_bf16(a1.v, b.v, acc1);
    }
}

__global__ void zero_f32(float* __restrict__ p, long n) {
    long i = (long)blockIdx.x * blockDim.x + threadIdx.x;
    long stride = (long)gridDim.x * blockDim.x;
    for (; i < n; i += stride) p[i] = 0.0f;
}

// Pack W[K,128] (f32 row-major) into WMMA B-operand tile layout (bf16).
// Rows k >= Ksrc zero-padded (gw1: 144 -> 160).
__global__ void pack_w(const float* __restrict__ W, unsigned short* __restrict__ dst,
                       int Ksrc, int Kpad) {
    int idx = blockIdx.x * 256 + threadIdx.x;
    int total = (Kpad / 32) * (HID / 16) * 512;
    if (idx >= total) return;
    int tile = idx >> 9;
    int lane = (idx >> 4) & 31;
    int i    = idx & 15;
    int kt = tile >> 3;
    int nt = tile & 7;
    int k = kt * 32 + ((lane >= 16) ? 16 : 0) + i;   // contiguous k order (same as A reads): lanes 16-31 hold K+16
    int n = nt * 16 + (lane & 15);
    float v = (k < Ksrc) ? W[k * HID + n] : 0.0f;
    dst[idx] = f2bf(v);
}

// ---------------- Edge model: 32 edges per block, 8 waves x 16 cols ----------------
__global__ void __launch_bounds__(256) edge_kernel(
    const float* __restrict__ x, const int* __restrict__ ei,
    const float* __restrict__ ea,
    const unsigned short* __restrict__ w1, const float* __restrict__ b1,
    const unsigned short* __restrict__ w2, const float* __restrict__ b2,
    float* __restrict__ agg_e) {
    __shared__ __attribute__((aligned(16))) unsigned short sIn[32 * 160];
    __shared__ __attribute__((aligned(16))) unsigned short sH[32 * 128];
    __shared__ int sCol[32];
    const int tid = threadIdx.x;
    const long e0 = (long)blockIdx.x * 32;
    unsigned* sInU = (unsigned*)sIn;

    for (int p = tid; p < 32 * 80; p += 256) {   // bf16 pairs
        int m = p / 80;
        int j = p - m * 80;
        int f = j * 2;
        long e = e0 + m;
        float a, b;
        if (f < 64)       { const float* s = x + (long)ei[e] * 64 + f;            a = s[0]; b = s[1]; }
        else if (f < 128) { const float* s = x + (long)ei[NE + e] * 64 + (f - 64); a = s[0]; b = s[1]; }
        else              { const float* s = ea + e * 32 + (f - 128);              a = s[0]; b = s[1]; }
        sInU[m * 80 + j] = pk2bf(a, b);
    }
    if (tid < 32) sCol[tid] = ei[NE + e0 + tid];
    __syncthreads();

    const int wave = tid >> 5;
    const int lane = tid & 31;
    const int ncol = wave * 16 + (lane & 15);
    const int rowb = (lane >> 4) << 3;           // C/D layout: lanes 16-31 hold M+8
    const v8f z = {0.f, 0.f, 0.f, 0.f, 0.f, 0.f, 0.f, 0.f};

    v8f acc0 = z, acc1 = z;
    mlp_layer2<5>(sIn, 160, w1 + wave * 512 + lane * 16, lane, acc0, acc1);
    float bias = b1[ncol];
#pragma unroll
    for (int r = 0; r < 8; ++r) {
        sH[(rowb + r) * 128 + ncol]      = f2bf(fmaxf(acc0[r] + bias, 0.f));
        sH[(16 + rowb + r) * 128 + ncol] = f2bf(fmaxf(acc1[r] + bias, 0.f));
    }
    __syncthreads();

    acc0 = z; acc1 = z;
    mlp_layer2<4>(sH, 128, w2 + wave * 512 + lane * 16, lane, acc0, acc1);
    float bias2 = b2[ncol];
#pragma unroll
    for (int r = 0; r < 8; ++r) {
        unsafeAtomicAdd(&agg_e[(long)sCol[rowb + r] * 128 + ncol],
                        fmaxf(acc0[r] + bias2, 0.f));
        unsafeAtomicAdd(&agg_e[(long)sCol[16 + rowb + r] * 128 + ncol],
                        fmaxf(acc1[r] + bias2, 0.f));
    }
}

// ---------------- Node model: 32 nodes per block ----------------
__global__ void __launch_bounds__(256) node_kernel(
    const float* __restrict__ x, const float* __restrict__ agg_e,
    const int* __restrict__ batch,
    const unsigned short* __restrict__ w1, const float* __restrict__ b1,
    const unsigned short* __restrict__ w2, const float* __restrict__ b2,
    float* __restrict__ agg_n) {
    __shared__ __attribute__((aligned(16))) unsigned short sIn[32 * 192];
    __shared__ __attribute__((aligned(16))) unsigned short sH[32 * 128];
    __shared__ int sB[32];
    const int tid = threadIdx.x;
    const long n0 = (long)blockIdx.x * 32;
    unsigned* sInU = (unsigned*)sIn;

    for (int p = tid; p < 32 * 96; p += 256) {
        int m = p / 96;
        int j = p - m * 96;
        int f = j * 2;
        long node = n0 + m;
        float a, b;
        if (f < 64) { const float* s = x + node * 64 + f;              a = s[0]; b = s[1]; }
        else        { const float* s = agg_e + node * 128 + (f - 64);  a = s[0]; b = s[1]; }
        sInU[m * 96 + j] = pk2bf(a, b);
    }
    if (tid < 32) sB[tid] = batch[n0 + tid];
    __syncthreads();

    const int wave = tid >> 5;
    const int lane = tid & 31;
    const int ncol = wave * 16 + (lane & 15);
    const int rowb = (lane >> 4) << 3;
    const v8f z = {0.f, 0.f, 0.f, 0.f, 0.f, 0.f, 0.f, 0.f};

    v8f acc0 = z, acc1 = z;
    mlp_layer2<6>(sIn, 192, w1 + wave * 512 + lane * 16, lane, acc0, acc1);
    float bias = b1[ncol];
#pragma unroll
    for (int r = 0; r < 8; ++r) {
        sH[(rowb + r) * 128 + ncol]      = f2bf(fmaxf(acc0[r] + bias, 0.f));
        sH[(16 + rowb + r) * 128 + ncol] = f2bf(fmaxf(acc1[r] + bias, 0.f));
    }
    __syncthreads();

    acc0 = z; acc1 = z;
    mlp_layer2<4>(sH, 128, w2 + wave * 512 + lane * 16, lane, acc0, acc1);
    float bias2 = b2[ncol];
#pragma unroll
    for (int r = 0; r < 8; ++r) {
        unsafeAtomicAdd(&agg_n[(long)sB[rowb + r] * 128 + ncol],
                        fmaxf(acc0[r] + bias2, 0.f));
        unsafeAtomicAdd(&agg_n[(long)sB[16 + rowb + r] * 128 + ncol],
                        fmaxf(acc1[r] + bias2, 0.f));
    }
}

// ---------------- Global model + final linear: 32 graphs per block ----------------
__global__ void __launch_bounds__(256) global_kernel(
    const float* __restrict__ u, const float* __restrict__ agg_n,
    const unsigned short* __restrict__ w1, const float* __restrict__ b1,
    const unsigned short* __restrict__ w2, const float* __restrict__ b2,
    const float* __restrict__ fw, const float* __restrict__ fb,
    float* __restrict__ out) {
    __shared__ __attribute__((aligned(16))) unsigned short sIn[32 * 160];
    __shared__ __attribute__((aligned(16))) unsigned short sH[32 * 128];
    __shared__ float sU[32 * 128];
    const int tid = threadIdx.x;
    const long g0 = (long)blockIdx.x * 32;
    unsigned* sInU = (unsigned*)sIn;

    for (int p = tid; p < 32 * 80; p += 256) {
        int m = p / 80;
        int j = p - m * 80;
        int f = j * 2;
        long g = g0 + m;
        float a, b;
        if (f < 16)       { const float* s = u + g * 16 + f;              a = s[0]; b = s[1]; }
        else if (f < 144) { const float* s = agg_n + g * 128 + (f - 16);  a = s[0]; b = s[1]; }
        else              { a = 0.f; b = 0.f; }   // K padded 144 -> 160 (weights zero-padded)
        sInU[m * 80 + j] = pk2bf(a, b);
    }
    __syncthreads();

    const int wave = tid >> 5;
    const int lane = tid & 31;
    const int ncol = wave * 16 + (lane & 15);
    const int rowb = (lane >> 4) << 3;
    const v8f z = {0.f, 0.f, 0.f, 0.f, 0.f, 0.f, 0.f, 0.f};

    v8f acc0 = z, acc1 = z;
    mlp_layer2<5>(sIn, 160, w1 + wave * 512 + lane * 16, lane, acc0, acc1);
    float bias = b1[ncol];
#pragma unroll
    for (int r = 0; r < 8; ++r) {
        sH[(rowb + r) * 128 + ncol]      = f2bf(fmaxf(acc0[r] + bias, 0.f));
        sH[(16 + rowb + r) * 128 + ncol] = f2bf(fmaxf(acc1[r] + bias, 0.f));
    }
    __syncthreads();

    acc0 = z; acc1 = z;
    mlp_layer2<4>(sH, 128, w2 + wave * 512 + lane * 16, lane, acc0, acc1);
    float bias2 = b2[ncol];
#pragma unroll
    for (int r = 0; r < 8; ++r) {
        sU[(rowb + r) * 128 + ncol]      = fmaxf(acc0[r] + bias2, 0.f);
        sU[(16 + rowb + r) * 128 + ncol] = fmaxf(acc1[r] + bias2, 0.f);
    }
    __syncthreads();

    if (tid < 32) {
        float s = fb[0];
        for (int n = 0; n < 128; ++n) s += sU[tid * 128 + n] * fw[n];
        out[g0 + tid] = s;
    }
}

extern "C" void kernel_launch(void* const* d_in, const int* in_sizes, int n_in,
                              void* d_out, int out_size, void* d_ws, size_t ws_size,
                              hipStream_t stream) {
    const float* x   = (const float*)d_in[0];
    const int*   ei  = (const int*)d_in[1];
    const float* ea  = (const float*)d_in[2];
    const float* u   = (const float*)d_in[3];
    const int*   bat = (const int*)d_in[4];
    const float* ew1 = (const float*)d_in[5];
    const float* eb1 = (const float*)d_in[6];
    const float* ew2 = (const float*)d_in[7];
    const float* eb2 = (const float*)d_in[8];
    const float* nw1 = (const float*)d_in[9];
    const float* nb1 = (const float*)d_in[10];
    const float* nw2 = (const float*)d_in[11];
    const float* nb2 = (const float*)d_in[12];
    const float* gw1 = (const float*)d_in[13];
    const float* gb1 = (const float*)d_in[14];
    const float* gw2 = (const float*)d_in[15];
    const float* gb2 = (const float*)d_in[16];
    const float* fw  = (const float*)d_in[17];
    const float* fb  = (const float*)d_in[18];
    float* out = (float*)d_out;

    // workspace layout
    float* agg_e = (float*)d_ws;                                  // 100000*128 f32
    float* agg_n = agg_e + (size_t)NN * HID;                      // 1024*128 f32
    unsigned short* ew1p = (unsigned short*)(agg_n + (size_t)NG * HID);
    unsigned short* ew2p = ew1p + 20480;   // 160K-pack
    unsigned short* nw1p = ew2p + 16384;   // 128K-pack
    unsigned short* nw2p = nw1p + 24576;   // 192K-pack
    unsigned short* gw1p = nw2p + 16384;   // 128K-pack
    unsigned short* gw2p = gw1p + 20480;   // 160K-pack (144 zero-padded)

    zero_f32<<<4096, 256, 0, stream>>>(agg_e, (long)NN * HID + (long)NG * HID);

    pack_w<<<80, 256, 0, stream>>>(ew1, ew1p, 160, 160);
    pack_w<<<64, 256, 0, stream>>>(ew2, ew2p, 128, 128);
    pack_w<<<96, 256, 0, stream>>>(nw1, nw1p, 192, 192);
    pack_w<<<64, 256, 0, stream>>>(nw2, nw2p, 128, 128);
    pack_w<<<80, 256, 0, stream>>>(gw1, gw1p, 144, 160);
    pack_w<<<64, 256, 0, stream>>>(gw2, gw2p, 128, 128);

    edge_kernel<<<NE / 32, 256, 0, stream>>>(x, ei, ea, ew1p, eb1, ew2p, eb2, agg_e);
    node_kernel<<<NN / 32, 256, 0, stream>>>(x, agg_e, bat, nw1p, nb1, nw2p, nb2, agg_n);
    global_kernel<<<NG / 32, 256, 0, stream>>>(u, agg_n, gw1p, gb1, gw2p, gb2, fw, fb, out);
}
